// Block_3453153706261
// MI455X (gfx1250) — matrix operand order repaired and checked by reading, not compiled, not run on hardware
//
#include <hip/hip_runtime.h>
#include <hip/hip_bf16.h>

// ---------------------------------------------------------------------------
// Transformer block for MI455X (gfx1250, wave32, WMMA).
// All matmuls run on v_wmma_f32_16x16x32_bf16 (bf16 in, f32 accumulate).
// Weights are pre-transposed (N-major) so every WMMA operand fragment is two
// contiguous global_load_b128 per lane -- no LDS staging, no block barriers
// in the GEMM hot loop. Attention is flash-style; V is pre-transposed too.
// ---------------------------------------------------------------------------

typedef __bf16 bf16;
typedef __attribute__((ext_vector_type(16))) __bf16 v16bf;
typedef __attribute__((ext_vector_type(8)))  float  v8f;

union AF { v16bf v; bf16  e[16]; uint4 q[2]; };  // operand fragment, 32B/lane
union CF { v8f   v; float e[8];  };              // accumulator fragment

#define WMMA_BF16(a, b, c) \
  __builtin_amdgcn_wmma_f32_16x16x32_bf16(false, (a), false, (b), (short)0, (c), false, false)

#define DIM     384
#define HEADS   6
#define DHEAD   64
#define SEQ     2048
#define BATCH   4
#define C3      1152   // 3*DIM
#define HIDDEN  1536
#define NTOK    (BATCH * SEQ)   // 8192

// ---------------------------------------------------------------------------
// Weight convert + transpose: src f32 [K][N] -> dst bf16 [N][K]
// ---------------------------------------------------------------------------
__global__ void cvt_transpose_bf16(const float* __restrict__ src, bf16* __restrict__ dst,
                                   int K, int N) {
  int i = blockIdx.x * blockDim.x + threadIdx.x;
  if (i >= K * N) return;
  int k = i / N, n = i % N;                 // consecutive threads: coalesced reads
  dst[(size_t)n * K + k] = (bf16)src[i];
}

// ---------------------------------------------------------------------------
// Repack V from qkv[token][C3] into vt[bh][d][n] (token-contiguous rows).
// ---------------------------------------------------------------------------
__global__ void repack_vt(const bf16* __restrict__ qkv, bf16* __restrict__ vt) {
  int i = blockIdx.x * blockDim.x + threadIdx.x;   // over B*H*DHEAD*SEQ
  int n  = i & (SEQ - 1);
  int t  = i >> 11;
  int d  = t & (DHEAD - 1);
  int bh = t >> 6;
  int b  = bh / HEADS, h = bh % HEADS;
  vt[i] = qkv[(size_t)(b * SEQ + n) * C3 + 2 * DIM + h * DHEAD + d];
}

// ---------------------------------------------------------------------------
// LayerNorm: one wave32 per token (384 elems -> 12 per lane), bf16 output.
// ---------------------------------------------------------------------------
__global__ __launch_bounds__(256) void ln_bf16_kernel(
    const float* __restrict__ x, const float* __restrict__ g,
    const float* __restrict__ b, bf16* __restrict__ out, int ntok)
{
  const int wave = threadIdx.x >> 5;
  const int lane = threadIdx.x & 31;
  const int tok  = blockIdx.x * (blockDim.x >> 5) + wave;
  if (tok >= ntok) return;
  const float* xr = x + (size_t)tok * DIM;
  float v[12], s = 0.f, s2 = 0.f;
#pragma unroll
  for (int j = 0; j < 12; j++) {
    float t = xr[lane + j * 32];
    v[j] = t; s += t; s2 += t * t;
  }
#pragma unroll
  for (int m = 16; m >= 1; m >>= 1) {
    s  += __shfl_xor(s,  m, 32);
    s2 += __shfl_xor(s2, m, 32);
  }
  const float mean = s * (1.0f / DIM);
  const float var  = s2 * (1.0f / DIM) - mean * mean;
  const float inv  = rsqrtf(var + 1e-5f);
  bf16* orow = out + (size_t)tok * DIM;
#pragma unroll
  for (int j = 0; j < 12; j++) {
    int c = lane + j * 32;
    orow[c] = (bf16)((v[j] - mean) * inv * g[c] + b[c]);
  }
}

// ---------------------------------------------------------------------------
// WMMA GEMM (TN): out[M,N] = A[M,K](bf16) @ BT[N,K](bf16)^T (+bias)(+gelu)(+res)
// Block = 256 threads (8 waves), block tile 128x64, wave tile 16x64.
// All operands: contiguous b128 global loads (weights are L2-resident).
// ---------------------------------------------------------------------------
__device__ __forceinline__ float gelu_tanh_f(float val) {
  float x3 = val * val * val;
  return 0.5f * val * (1.0f + tanhf(0.7978845608028654f * (val + 0.044715f * x3)));
}

__global__ __launch_bounds__(256) void wmma_gemm_tn_kernel(
    const bf16* __restrict__ A,    // [M][K]
    const bf16* __restrict__ BT,   // [N][K]
    const float* __restrict__ bias, const float* __restrict__ residual,
    float* __restrict__ outF, bf16* __restrict__ outH,
    int M, int N, int K, int act_gelu)
{
  const int wave = threadIdx.x >> 5;
  const int lane = threadIdx.x & 31;
  const int half = lane >> 4;
  const int l16  = lane & 15;
  const int m0   = blockIdx.y * 128 + wave * 16;
  const int n0   = blockIdx.x * 64;

  CF acc[4];
#pragma unroll
  for (int f = 0; f < 4; f++)
#pragma unroll
    for (int i = 0; i < 8; i++) acc[f].e[i] = 0.f;

  const bf16* arow = A + (size_t)(m0 + l16) * K + 16 * half;
  const bf16* brow0 = BT + (size_t)(n0 + l16) * K + 16 * half;

#pragma unroll 4
  for (int k0 = 0; k0 < K; k0 += 32) {
    AF a;
    a.q[0] = *(const uint4*)(arow + k0);
    a.q[1] = *(const uint4*)(arow + k0 + 8);
#pragma unroll
    for (int f = 0; f < 4; f++) {
      const bf16* bp = brow0 + (size_t)(f * 16) * K + k0;
      AF bfr;
      bfr.q[0] = *(const uint4*)(bp);
      bfr.q[1] = *(const uint4*)(bp + 8);
      acc[f].v = WMMA_BF16(a.v, bfr.v, acc[f].v);
    }
  }

  // epilogue
#pragma unroll
  for (int f = 0; f < 4; f++) {
    int col = n0 + f * 16 + l16;
    float bv = bias ? bias[col] : 0.f;
#pragma unroll
    for (int v = 0; v < 8; v++) {
      int r = m0 + v + 8 * half;
      float val = acc[f].e[v] + bv;
      if (act_gelu) val = gelu_tanh_f(val);
      size_t idx = (size_t)r * N + col;
      if (residual) val += residual[idx];
      if (outF) outF[idx] = val;
      if (outH) outH[idx] = (bf16)val;
    }
  }
}

// ---------------------------------------------------------------------------
// Flash attention. grid = (SEQ/64, BATCH*HEADS), block = 128 (4 waves).
// Each wave: 16 query rows. Q/K frags: contiguous b128 from qkv.
// V frags: contiguous b128 from pre-transposed vt[bh][d][n].
// Only LDS use: per-wave P tile transpose (C-layout -> A-layout).
// ---------------------------------------------------------------------------
__global__ __launch_bounds__(128) void attn_kernel(
    const bf16* __restrict__ qkv, const bf16* __restrict__ vt,
    bf16* __restrict__ o)
{
  __shared__ bf16 pl[4][16 * 32];   // per-wave P tile (1 KB each)

  const int wave = threadIdx.x >> 5;
  const int lane = threadIdx.x & 31;
  const int half = lane >> 4;
  const int l16  = lane & 15;
  const int bh   = blockIdx.y;
  const int bb   = bh / HEADS;
  const int hh   = bh % HEADS;
  const int m0   = blockIdx.x * 64 + wave * 16;
  const int qoff = hh * DHEAD;
  const int koff = DIM + hh * DHEAD;
  const size_t base = (size_t)bb * SEQ * C3;
  const bf16* vbase = vt + (size_t)bh * DHEAD * SEQ;   // vt[bh][d][n]

  // Q fragments (16x64 split into two K=32 A-frags): contiguous 16B loads
  AF qa[2];
  {
    const bf16* qrow = qkv + base + (size_t)(m0 + l16) * C3 + qoff + 16 * half;
#pragma unroll
    for (int h2 = 0; h2 < 2; h2++) {
      qa[h2].q[0] = *(const uint4*)(qrow + h2 * 32);
      qa[h2].q[1] = *(const uint4*)(qrow + h2 * 32 + 8);
    }
  }

  CF acc[4];
#pragma unroll
  for (int f = 0; f < 4; f++)
#pragma unroll
    for (int i = 0; i < 8; i++) acc[f].e[i] = 0.f;

  float mrow[8], lrow[8];
#pragma unroll
  for (int v = 0; v < 8; v++) { mrow[v] = -3.0e38f; lrow[v] = 0.f; }

  CF zero;
#pragma unroll
  for (int i = 0; i < 8; i++) zero.e[i] = 0.f;

  const float SC = 0.18033688011112042f;   // (1/sqrt(64)) * log2(e)

  for (int kc = 0; kc < SEQ; kc += 32) {
    // ---- S = Q @ K^T : K^T fragment is contiguous in the feature dim ----
    CF s[2];
#pragma unroll
    for (int f = 0; f < 2; f++) {
      const bf16* krow = qkv + base + (size_t)(kc + f * 16 + l16) * C3 + koff + 16 * half;
      AF bk;
      bk.q[0] = *(const uint4*)(krow);
      bk.q[1] = *(const uint4*)(krow + 8);
      s[f].v = WMMA_BF16(qa[0].v, bk.v, zero.v);
      bk.q[0] = *(const uint4*)(krow + 32);
      bk.q[1] = *(const uint4*)(krow + 40);
      s[f].v = WMMA_BF16(qa[1].v, bk.v, s[f].v);
    }

    // ---- online softmax in exp2 domain (scale folded into log2 factor) ----
    float alpha[8];
#pragma unroll
    for (int v = 0; v < 8; v++) {
      float s0 = s[0].e[v] * SC;
      float s1 = s[1].e[v] * SC;
      float mx = fmaxf(s0, s1);
#pragma unroll
      for (int msk = 8; msk >= 1; msk >>= 1) mx = fmaxf(mx, __shfl_xor(mx, msk, 32));
      float nm = fmaxf(mrow[v], mx);
      float al = exp2f(mrow[v] - nm);
      float p0 = exp2f(s0 - nm);
      float p1 = exp2f(s1 - nm);
      float rs = p0 + p1;
#pragma unroll
      for (int msk = 8; msk >= 1; msk >>= 1) rs += __shfl_xor(rs, msk, 32);
      lrow[v] = lrow[v] * al + rs;
      mrow[v] = nm;
      alpha[v] = al;
      s[0].e[v] = p0; s[1].e[v] = p1;
    }
#pragma unroll
    for (int f = 0; f < 4; f++)
#pragma unroll
      for (int v = 0; v < 8; v++) acc[f].e[v] *= alpha[v];

    // ---- P: C-layout frag -> row-major 16x32 tile in per-wave LDS ----
#pragma unroll
    for (int f = 0; f < 2; f++)
#pragma unroll
      for (int v = 0; v < 8; v++)
        pl[wave][(v + 8 * half) * 32 + f * 16 + l16] = (bf16)s[f].e[v];
    __builtin_amdgcn_wave_barrier();      // DS is in-order per wave; order compiler

    AF pa;
    pa.q[0] = *(const uint4*)(&pl[wave][l16 * 32 + 16 * half]);
    pa.q[1] = *(const uint4*)(&pl[wave][l16 * 32 + 8 + 16 * half]);
    __builtin_amdgcn_wave_barrier();

    // ---- O += P @ V : V fragment contiguous from vt[bh][d][n] ----
#pragma unroll
    for (int f2 = 0; f2 < 4; f2++) {
      const bf16* vp = vbase + (size_t)(f2 * 16 + l16) * SEQ + kc + 16 * half;
      AF bv;
      bv.q[0] = *(const uint4*)(vp);
      bv.q[1] = *(const uint4*)(vp + 8);
      acc[f2].v = WMMA_BF16(pa.v, bv.v, acc[f2].v);
    }
  }

  // ---- epilogue: normalize and store o[b,n, h*64 + d] as bf16 ----
#pragma unroll
  for (int v = 0; v < 8; v++) {
    float inv = 1.0f / lrow[v];
    int r = m0 + v + 8 * half;
#pragma unroll
    for (int f2 = 0; f2 < 4; f2++)
      o[((size_t)bb * SEQ + r) * DIM + hh * DHEAD + f2 * 16 + l16] =
          (bf16)(acc[f2].e[v] * inv);
  }
}

// ---------------------------------------------------------------------------
// Launcher
// ---------------------------------------------------------------------------
extern "C" void kernel_launch(void* const* d_in, const int* in_sizes, int n_in,
                              void* d_out, int out_size, void* d_ws, size_t ws_size,
                              hipStream_t stream)
{
  const float* x      = (const float*)d_in[0];
  const float* ln1_g  = (const float*)d_in[1];
  const float* ln1_b  = (const float*)d_in[2];
  const float* w_qkv  = (const float*)d_in[3];
  const float* w_proj = (const float*)d_in[4];
  const float* b_proj = (const float*)d_in[5];
  const float* ln2_g  = (const float*)d_in[6];
  const float* ln2_b  = (const float*)d_in[7];
  const float* w_fc1  = (const float*)d_in[8];
  const float* b_fc1  = (const float*)d_in[9];
  const float* w_fc2  = (const float*)d_in[10];
  const float* b_fc2  = (const float*)d_in[11];
  float* out = (float*)d_out;

  char* ws = (char*)d_ws;
  size_t off = 0;
  auto carve = [&](size_t bytes) -> char* {
    char* p = ws + off;
    off = (off + bytes + 255) & ~(size_t)255;
    return p;
  };
  bf16*  wqkvT  = (bf16*)carve((size_t)DIM * C3 * 2);       // [1152][384]
  bf16*  wprojT = (bf16*)carve((size_t)DIM * DIM * 2);      // [384][384]
  bf16*  wfc1T  = (bf16*)carve((size_t)DIM * HIDDEN * 2);   // [1536][384]
  bf16*  wfc2T  = (bf16*)carve((size_t)HIDDEN * DIM * 2);   // [384][1536]
  bf16*  h_h    = (bf16*)carve((size_t)NTOK * DIM * 2);     // ln1 / ln2 output
  bf16*  qkv_h  = (bf16*)carve((size_t)NTOK * C3 * 2);
  bf16*  vt_h   = (bf16*)carve((size_t)BATCH * HEADS * DHEAD * SEQ * 2);
  bf16*  o_h    = (bf16*)carve((size_t)NTOK * DIM * 2);
  bf16*  hfc_h  = (bf16*)carve((size_t)NTOK * HIDDEN * 2);
  float* x1     = (float*)carve((size_t)NTOK * DIM * 4);    // post-attn residual
  (void)ws_size; (void)in_sizes; (void)n_in; (void)out_size;

  // 1) weights -> bf16, transposed to [N][K]
  {
    int n;
    n = DIM * C3;     cvt_transpose_bf16<<<(n + 255) / 256, 256, 0, stream>>>(w_qkv,  wqkvT,  DIM, C3);
    n = DIM * DIM;    cvt_transpose_bf16<<<(n + 255) / 256, 256, 0, stream>>>(w_proj, wprojT, DIM, DIM);
    n = DIM * HIDDEN; cvt_transpose_bf16<<<(n + 255) / 256, 256, 0, stream>>>(w_fc1,  wfc1T,  DIM, HIDDEN);
    n = HIDDEN * DIM; cvt_transpose_bf16<<<(n + 255) / 256, 256, 0, stream>>>(w_fc2,  wfc2T,  HIDDEN, DIM);
  }

  // 2) h = LN1(x)
  ln_bf16_kernel<<<NTOK / 8, 256, 0, stream>>>(x, ln1_g, ln1_b, h_h, NTOK);

  // 3) qkv = h @ w_qkv                     [8192 x 1152]
  wmma_gemm_tn_kernel<<<dim3(C3 / 64, NTOK / 128), 256, 0, stream>>>(
      h_h, wqkvT, nullptr, nullptr, nullptr, qkv_h, NTOK, C3, DIM, 0);

  // 3b) vt[bh][d][n] = V^T
  repack_vt<<<(BATCH * HEADS * DHEAD * SEQ) / 256, 256, 0, stream>>>(qkv_h, vt_h);

  // 4) o = attention(qkv)                  [8192 x 384]
  attn_kernel<<<dim3(SEQ / 64, BATCH * HEADS), 128, 0, stream>>>(qkv_h, vt_h, o_h);

  // 5) x1 = x + o @ w_proj + b_proj
  wmma_gemm_tn_kernel<<<dim3(DIM / 64, NTOK / 128), 256, 0, stream>>>(
      o_h, wprojT, b_proj, x, x1, nullptr, NTOK, DIM, DIM, 0);

  // 6) h = LN2(x1)
  ln_bf16_kernel<<<NTOK / 8, 256, 0, stream>>>(x1, ln2_g, ln2_b, h_h, NTOK);

  // 7) hfc = gelu(h @ w_fc1 + b_fc1)       [8192 x 1536]
  wmma_gemm_tn_kernel<<<dim3(HIDDEN / 64, NTOK / 128), 256, 0, stream>>>(
      h_h, wfc1T, b_fc1, nullptr, nullptr, hfc_h, NTOK, HIDDEN, DIM, 1);

  // 8) out = x1 + hfc @ w_fc2 + b_fc2
  wmma_gemm_tn_kernel<<<dim3(DIM / 64, NTOK / 128), 256, 0, stream>>>(
      hfc_h, wfc2T, b_fc2, x1, out, nullptr, NTOK, DIM, HIDDEN, 0);
}
